// LengthConverter_71021579207174
// MI455X (gfx1250) — matrix operand order repaired and checked by reading, not compiled, not run on hardware
//
#include <hip/hip_runtime.h>

// MI455X / gfx1250, wave32. Fused length-converter:
//   weight = softmax_s( -(2^clip(s-mu,±100))/(2σ²), masked )  generated on the fly in
//   WMMA A-register layout, z' = weight @ z via v_wmma_f32_16x16x32_bf16 (f32 accum).
// Roofline: 206 GFLOP GEMM, ~320MB HBM floor (~14us @23.3TB/s); z (128MB) lives in 192MB L2.
// Block tile 128x128, K-step 32, 8 waves (one 16-row M-slice each, all 128 N cols),
// so no duplicated weight generation inside a block and 8 WMMAs per A-fragment.

typedef __attribute__((ext_vector_type(16))) __bf16 v16bf;
typedef __attribute__((ext_vector_type(8)))  __bf16 v8bf;
typedef __attribute__((ext_vector_type(8)))  float  v8f;
typedef __attribute__((ext_vector_type(4)))  float  v4f;

constexpr int B_   = 16;
constexpr int LX   = 2048;
constexpr int H_   = 1024;
constexpr int LOUT = 3072;

constexpr int BM = 128;   // block tile M (L_out rows) = 8 waves x 16
constexpr int BN = 128;   // block tile N (H cols)     = 8 fragments per wave
constexpr int BK = 32;    // K step (= bf16 WMMA K)
constexpr int KSTR = 40;  // LDS halves per column (80B: 16B-aligned b128 frag loads)
constexpr float LOG2E = 1.4426950408889634f;

// ---------------- kernel 1: n[b] = sum(z_mask[b,:]) ----------------
__global__ __launch_bounds__(256) void nsum_kernel(const float* __restrict__ zmask,
                                                   float* __restrict__ nW) {
  __shared__ float red[256];
  const int b = blockIdx.x;
  float s = 0.f;
  for (int i = threadIdx.x; i < LX; i += 256) s += zmask[(size_t)b * LX + i];
  red[threadIdx.x] = s;
  __syncthreads();
  for (int off = 128; off > 0; off >>= 1) {
    if (threadIdx.x < off) red[threadIdx.x] += red[threadIdx.x + off];
    __syncthreads();
  }
  if (threadIdx.x == 0) nW[b] = red[0];
}

// ------- kernel 2: softmax denominators D[b,t] + z'_mask output -------
__global__ __launch_bounds__(256) void denom_kernel(const long long* __restrict__ ls,
                                                    const float* __restrict__ sigma,
                                                    const float* __restrict__ nW,
                                                    float* __restrict__ D,
                                                    float* __restrict__ outMask) {
  const int wid  = threadIdx.x >> 5;
  const int lane = threadIdx.x & 31;
  const int row  = blockIdx.x * 8 + wid;          // row in [0, B*LOUT)
  const int b = row / LOUT;
  const int t = row - b * LOUT;
  const float sg = sigma[0];
  const float cs = LOG2E / (2.f * sg * sg);       // exp(logit) = exp2(-2^dist * cs)
  const float nb = nW[b];
  const long long lsv = ls[b];
  const float mu = (float)t * nb / (float)lsv;

  float sum = 0.f;
  for (int s = lane; s < LX; s += 32) {
    const float fs = (float)s;
    if (fs < nb) {                                // valid source position
      const float dist = fminf(fmaxf(fs - mu, -100.f), 100.f);
      sum += __builtin_amdgcn_exp2f(-__builtin_amdgcn_exp2f(dist) * cs);
    }
  }
#pragma unroll
  for (int m = 16; m > 0; m >>= 1) sum += __shfl_xor(sum, m, 32);
  if (lane == 0) {
    D[row] = sum;                                  // >= ~0.6 always, no max-sub needed
    outMask[row] = ((long long)t < lsv) ? 1.f : 0.f;
  }
}

// ---------------- kernel 3: fused weight-gen + bf16 WMMA GEMM ----------------
__global__ __launch_bounds__(256) void lconv_gemm_kernel(const float* __restrict__ z,
                                                         const long long* __restrict__ ls,
                                                         const float* __restrict__ sigma,
                                                         const float* __restrict__ nW,
                                                         const float* __restrict__ D,
                                                         float* __restrict__ out) {
  __shared__ __bf16 zt[2][BN * KSTR];             // z tile, [col][k] bf16, double-buffered (20KB)

  const int b   = blockIdx.z;
  const int bm0 = blockIdx.y * BM;
  const int h0  = blockIdx.x * BN;
  const int tid = threadIdx.x;
  const int lane = tid & 31;
  const int wm   = tid >> 5;                      // wave id == M sub-tile (8 x 16 rows)

  const float sg  = sigma[0];
  const float cs  = LOG2E / (2.f * sg * sg);
  const float nb  = nW[b];
  const long long lsv = ls[b];

  // A-matrix row handled by this lane (16x32 bf16 A layout: M = lane & 15)
  const int   t_row = bm0 + wm * 16 + (lane & 15);
  const float mu    = (float)t_row * nb / (float)lsv;
  const float invD  = 1.f / D[(size_t)b * LOUT + t_row];

  const float* zb = z + (size_t)b * LX * H_;

  // ---- staging: 32(K) x 128(N) f32 tile -> bf16 [col][k] in LDS ----
  // 512 units of (k-pair, 4-col group); 2 units per thread; float4 global loads.
  v4f ra[2], rb[2];
  auto load_phase = [&](int k0) {
#pragma unroll
    for (int u = 0; u < 2; ++u) {
      const int unit = tid + u * 256;
      const int cg   = unit & 31;                 // col group (4 cols)
      const int kk   = (unit >> 5) * 2;           // k pair base
      const float* gp = zb + (size_t)(k0 + kk) * H_ + (h0 + cg * 4);
      ra[u] = *(const v4f*)gp;                    // global_load_b128, row kk
      rb[u] = *(const v4f*)(gp + H_);             // global_load_b128, row kk+1
    }
  };
  auto store_phase = [&](int buf) {
#pragma unroll
    for (int u = 0; u < 2; ++u) {
      const int unit = tid + u * 256;
      const int c    = (unit & 31) * 4;
      const int kk   = (unit >> 5) * 2;
#pragma unroll
      for (int i = 0; i < 4; ++i) {
        union { unsigned w; __bf16 h[2]; } pk;
        pk.h[0] = (__bf16)ra[u][i];
        pk.h[1] = (__bf16)rb[u][i];
        *(unsigned*)&zt[buf][(c + i) * KSTR + kk] = pk.w;   // transposed [N][K] tile
      }
    }
  };

  v8f acc[8] = {};                                // 8 N-fragments of 16x16 f32

  load_phase(0);
  store_phase(0);
  __syncthreads();

  const int koffA  = (lane >= 16) ? 16 : 0;       // A k order used here (contiguous): lanes 16-31 hold K=16..31
  const int khalfB = (lane >> 4) * 16;            // B k order used here (contiguous): lanes 16-31 hold K=16..31
  const int colB   = lane & 15;

  union Bv { v16bf v; v8bf p[2]; };
  auto loadB = [&](Bv& d, int buf, int f) {
    const __bf16* bp = &zt[buf][(f * 16 + colB) * KSTR + khalfB];
    d.p[0] = *(const v8bf*)(bp);                  // ds_load_b128
    d.p[1] = *(const v8bf*)(bp + 8);              // ds_load_b128
  };

  for (int k0 = 0, it = 0; k0 < LX; k0 += BK, ++it) {
    const int  buf  = it & 1;
    const bool more = (k0 + BK) < LX;
    if (more) load_phase(k0 + BK);                // prefetch next tile into regs

    // ---- A fragment: weights computed directly in WMMA A-register layout ----
    union { v16bf v; __bf16 h[16]; } a;
#pragma unroll
    for (int i = 0; i < 16; ++i) {
      const int   K    = i + koffA;
      const float fs   = (float)(k0 + K);
      const float dist = fminf(fmaxf(fs - mu, -100.f), 100.f);
      const float e    = (fs < nb)
          ? __builtin_amdgcn_exp2f(-__builtin_amdgcn_exp2f(dist) * cs) * invD
          : 0.f;
      a.h[i] = (__bf16)e;
    }

    // ---- 8 WMMAs over 128 columns, one-fragment B lookahead ----
    Bv cur, nxt;
    loadB(cur, buf, 0);
#pragma unroll
    for (int f = 0; f < 8; ++f) {
      if (f < 7) loadB(nxt, buf, f + 1);
      acc[f] = __builtin_amdgcn_wmma_f32_16x16x32_bf16(
          false, a.v, false, cur.v, (short)0, acc[f], false, false);
      if (f < 7) cur = nxt;
    }

    if (more) store_phase(buf ^ 1);
    __syncthreads();
  }

  // ---- epilogue: C/D layout M = r + 8*(lane>=16), N = lane&15; apply row mask ----
  const int mbase = bm0 + wm * 16 + ((lane >> 4) << 3);
  const int nbase = h0 + (lane & 15);
#pragma unroll
  for (int f = 0; f < 8; ++f) {
#pragma unroll
    for (int r = 0; r < 8; ++r) {
      const int tt = mbase + r;
      const float keep = ((long long)tt < lsv) ? 1.f : 0.f;
      out[((size_t)b * LOUT + tt) * H_ + (nbase + f * 16)] = acc[f][r] * keep;
    }
  }
}

extern "C" void kernel_launch(void* const* d_in, const int* in_sizes, int n_in,
                              void* d_out, int out_size, void* d_ws, size_t ws_size,
                              hipStream_t stream) {
  const float*     z     = (const float*)d_in[0];      // [B, LX, H] f32
  const long long* ls    = (const long long*)d_in[1];  // [B] int64
  const float*     zmask = (const float*)d_in[2];      // [B, LX] f32
  const float*     sigma = (const float*)d_in[3];      // scalar f32

  float* out     = (float*)d_out;                      // z' then z'_mask
  float* nW      = (float*)d_ws;                       // [B]
  float* Dden    = nW + B_;                            // [B*LOUT]
  float* outMask = out + (size_t)B_ * LOUT * H_;

  nsum_kernel<<<B_, 256, 0, stream>>>(zmask, nW);
  denom_kernel<<<(B_ * LOUT) / 8, 256, 0, stream>>>(ls, sigma, nW, Dden, outMask);
  lconv_gemm_kernel<<<dim3(H_ / BN, LOUT / BM, B_), 256, 0, stream>>>(
      z, ls, sigma, nW, Dden, out);
}
